// DelayRNN_47751446397284
// MI455X (gfx1250) — matrix operand order repaired and checked by reading, not compiled, not run on hardware
//
#include <hip/hip_runtime.h>
#include <hip/hip_bf16.h>
#include <math.h>

// ---------------------------------------------------------------------------
// DelayRNN for MI455X (gfx1250, wave32, WMMA).
//
// 1 workgroup (256 threads = 8 wave32) per batch element (batches are fully
// independent). The 64KB delay-line state lives in LDS for the whole sequence
// (rotating base pointer; the zero-pad slot is the rotated-out head slot).
// All four weight matrices are stationary f16 WMMA B-fragments in VGPRs.
// GEMVs: v_wmma_f32_16x16x32_f16 with the activation vector in row 0 of A,
// split into TWO independent accumulator chains (even/odd K-tiles) so the
// ds_load_b128 pairs overlap with WMMA execution instead of serializing.
// 4 barriers/step; decoder output GEMV overlaps the h1 phase on waves 4-7;
// next-step staging is folded into the blend phase.
// ---------------------------------------------------------------------------

typedef __attribute__((ext_vector_type(16))) _Float16 v16h;
typedef __attribute__((ext_vector_type(8)))  _Float16 v8h;
typedef __attribute__((ext_vector_type(8)))  float    v8f;
typedef __attribute__((ext_vector_type(4)))  float    v4f;

#define MAX_DELAY 128
#define INPUT     64
#define HIDDEN    128
#define HALF_     64
#define OUTD      64
#define BATCH     32
#define TLEN      128
#define OUT_LEN   64
#define KCAT      (INPUT + HIDDEN)   // 192
#define NTHREADS  256

// LDS layout (bytes): buffer + small activation staging (~66.3 KB of 320 KB).
#define OFF_BUF    0        // 128*128 f32        = 65536
#define OFF_MSG    65536    // 128 f32            = 512
#define OFF_TAU    66048    // 128 f32            = 512
#define OFF_MEMI   66560    // 128 f32            = 512
#define OFF_A16    67072    // 192 f16            = 384
#define OFF_MSG16  67456    // 128 f16            = 256
#define OFF_H16    67712    // 64  f16            = 128
#define OFF_ZERO   67840    // 32  f16 zeros      = 64
#define SMEM_BYTES 67904

// Native-instruction activations (v_exp_f32 / v_rcp_f32): these sit on the
// serial recurrence chain, so avoid the branchy libm expansions.
__device__ __forceinline__ float fast_sigmoid(float z) {
  float e = __builtin_amdgcn_exp2f(z * -1.4426950408889634f);   // e^-z
  return __builtin_amdgcn_rcpf(1.0f + e);
}
__device__ __forceinline__ float fast_tanh(float z) {
  float e = __builtin_amdgcn_exp2f(z * 2.8853900817779268f);    // e^(2z)
  return 1.0f - 2.0f * __builtin_amdgcn_rcpf(1.0f + e);         // saturates +-1
}

// Load one 32x16 (KxN) f16 B-fragment from a row-major f32 weight matrix.
// CDNA5 16-bit B layout: lane&15 = column; lanes>=16 carry K=8..15,24..31 of
// the tile; v16h element i <-> K = kbase + i (i<8), kbase + 8 + i (i>=8).
__device__ __forceinline__ v16h load_B_frag(const float* __restrict__ W,
                                            int ldn, int kt, int nt, int lane) {
  int n  = nt * 16 + (lane & 15);
  int kb = kt * 32 + ((lane & 16) ? 8 : 0);
  v16h Bv;
#pragma unroll
  for (int i = 0; i < 16; ++i)
    Bv[i] = (_Float16)W[(size_t)(kb + i + (i >= 8 ? 8 : 0)) * ldn + n];
  return Bv;
}

__device__ __forceinline__ v16h make_A(const _Float16* src) {
  v8h c0 = *(const v8h*)(src);        // K = kb+sub    .. +7   (16B, b128)
  v8h c1 = *(const v8h*)(src + 16);   // K = kb+sub+16 .. +23  (16B, b128)
  return __builtin_shufflevector(c0, c1, 0, 1, 2, 3, 4, 5, 6, 7,
                                         8, 9, 10, 11, 12, 13, 14, 15);
}

// GEMV via WMMA: activation vector (LDS f16, 16B-aligned) occupies row 0 of A
// (16-bit A layout: lane 0 holds K=0..7,16..23; lane 16 holds K=8..15,24..31),
// rows 1..15 read a zero block (branch-free pointer select). K-tiles are
// split across TWO independent accumulator chains so loads and WMMAs overlap
// (the single-chain version forced s_wait_dscnt 0 before every WMMA).
// Result (lanes<16): dot(vec, W[:, n]).  NKT must be even.
template <int NKT>
__device__ __forceinline__ float vec_wmma(const _Float16* __restrict__ vec,
                                          const _Float16* __restrict__ zro,
                                          const v16h* Bf, int lane) {
  const bool row0 = (lane & 15) == 0;
  const int  sub  = (lane & 16) ? 8 : 0;
  v8f acc0 = {}, acc1 = {};
#pragma unroll
  for (int kt = 0; kt < NKT; kt += 2) {
    const _Float16* s0 = row0 ? (vec + (kt + 0) * 32 + sub) : zro;
    const _Float16* s1 = row0 ? (vec + (kt + 1) * 32 + sub) : zro;
    v16h A0 = make_A(s0);
    v16h A1 = make_A(s1);
    acc0 = __builtin_amdgcn_wmma_f32_16x16x32_f16(false, A0, false, Bf[kt + 0],
                                                  (short)0, acc0, false, false);
    acc1 = __builtin_amdgcn_wmma_f32_16x16x32_f16(false, A1, false, Bf[kt + 1],
                                                  (short)0, acc1, false, false);
  }
  return acc0[0] + acc1[0];
}

extern "C" __global__ void __launch_bounds__(NTHREADS, 1)
delay_rnn_kernel(const float* __restrict__ x,   const int* __restrict__ lengths,
                 const float* __restrict__ Wm,  const float* __restrict__ bm,
                 const float* __restrict__ W1,  const float* __restrict__ b1,
                 const float* __restrict__ W2,  const float* __restrict__ b2,
                 const float* __restrict__ Wo,  const float* __restrict__ bo,
                 float* __restrict__ out) {
  extern __shared__ __align__(16) char smem[];
  float*    sBuf   = (float*)(smem + OFF_BUF);     // [HIDDEN][MAX_DELAY]
  float*    sMsg   = (float*)(smem + OFF_MSG);
  float*    sTau   = (float*)(smem + OFF_TAU);
  float*    sMemI  = (float*)(smem + OFF_MEMI);
  _Float16* a16    = (_Float16*)(smem + OFF_A16);  // [x_t ; head]  (192)
  _Float16* sMsg16 = (_Float16*)(smem + OFF_MSG16);
  _Float16* sH16   = (_Float16*)(smem + OFF_H16);
  _Float16* sZero  = (_Float16*)(smem + OFF_ZERO); // 32 zero halves

  const int tid  = threadIdx.x;
  const int lane = tid & 31;
  const int wave = tid >> 5;
  const int b    = blockIdx.x;

  // ---- Stationary weights: f16 WMMA B-fragments in VGPRs (loaded once) ----
  v16h msgB[6];   // Wm (192x128): this wave's N-tile = wave, kt = 0..5
  v16h tgB[4];    // W2 (64x256) : N-tiles wave*2, wave*2+1 ; kt = 0..1
  v16h auxB[4];   // waves 0-3: W1 (128x64) ; waves 4-7: Wo (128x64)
#pragma unroll
  for (int kt = 0; kt < 6; ++kt) msgB[kt] = load_B_frag(Wm, HIDDEN, kt, wave, lane);
  // tgB interleaved so vec_wmma's chain pairing matches: [nt0 kt0, nt0 kt1]
  // and [nt1 kt0, nt1 kt1] each form a 2-tile GEMV.
#pragma unroll
  for (int r = 0; r < 2; ++r)
#pragma unroll
    for (int kt = 0; kt < 2; ++kt)
      tgB[r * 2 + kt] = load_B_frag(W2, 2 * HIDDEN, kt, wave * 2 + r, lane);
  if (wave < 4) {
#pragma unroll
    for (int kt = 0; kt < 4; ++kt) auxB[kt] = load_B_frag(W1, HALF_, kt, wave, lane);
  } else {
#pragma unroll
    for (int kt = 0; kt < 4; ++kt) auxB[kt] = load_B_frag(Wo, OUTD, kt, wave - 4, lane);
  }
  const float bm_r   = bm[wave * 16 + (lane & 15)];
  const float b2_r0  = b2[(wave * 2 + 0) * 16 + (lane & 15)];
  const float b2_r1  = b2[(wave * 2 + 1) * 16 + (lane & 15)];
  const float baux_r = (wave < 4) ? b1[wave * 16 + (lane & 15)]
                                  : bo[(wave - 4) * 16 + (lane & 15)];

  // Zero delay buffer + zero block (fresh state every launch: deterministic).
  for (int i = tid; i < HIDDEN * MAX_DELAY; i += NTHREADS) sBuf[i] = 0.0f;
  if (tid < 32) sZero[tid] = (_Float16)0.0f;

  int len = lengths[b];
  if (len < 0) len = 0;
  if (len > TLEN) len = TLEN;

  int base = 0;  // rotating head: logical L[j] = sBuf[h][(base+j)&127]

  // Initial staging: a = [x_0 (or zeros if len==0) ; head of zeroed buffer].
  if (tid < INPUT) {
    a16[tid] = (len > 0) ? (_Float16)x[(size_t)b * TLEN * INPUT + tid]
                         : (_Float16)0.0f;
  } else if (tid < KCAT) {
    a16[tid] = (_Float16)0.0f;
  }
  __syncthreads();

  // One recurrent step (a16 already staged). xnext = next step's x row
  // (nullptr -> zeros, which is also the decoder's input). emit -> write
  // msg @ Wo + bo for decoder step s (runs on waves 4-7 during the h1 phase).
  auto step = [&](const float* xnext, bool emit, int s) {
    // --- msg = tanh([x;head] @ Wm + bm)      (8 waves x 6 WMMA)
    if (tid == 0 && xnext) __builtin_prefetch(xnext, 0, 1);
    {
      float r = vec_wmma<6>(a16, sZero, msgB, lane);
      if (lane < 16) {
        int n = wave * 16 + lane;
        float m = fast_tanh(r + bm_r);
        sMsg[n] = m;
        sMsg16[n] = (_Float16)m;
      }
    }
    __syncthreads();
    // --- h = relu(msg @ W1 + b1) on waves 0-3; decoder out GEMV on waves 4-7
    if (wave < 4) {
      float r = vec_wmma<4>(sMsg16, sZero, auxB, lane);
      if (lane < 16) {
        sH16[wave * 16 + lane] = (_Float16)fmaxf(r + baux_r, 0.0f);
      }
    } else if (emit) {
      float r = vec_wmma<4>(sMsg16, sZero, auxB, lane);
      if (lane < 16) {
        int n = (wave - 4) * 16 + lane;
        out[((size_t)b * OUT_LEN + s) * OUTD + n] = r + baux_r;
      }
    }
    __syncthreads();
    // --- tg = sigmoid(h @ W2 + b2); tau = tg[:128], mem_i = tg[128:]
    {
      float r0 = vec_wmma<2>(sH16, sZero, tgB, lane);
      float r1 = vec_wmma<2>(sH16, sZero, tgB + 2, lane);
      if (lane < 16) {
        int n0 = (wave * 2 + 0) * 16 + lane;
        int n1 = (wave * 2 + 1) * 16 + lane;
        float s0 = fast_sigmoid(r0 + b2_r0);
        float s1 = fast_sigmoid(r1 + b2_r1);
        if (n0 < HIDDEN) sTau[n0] = s0; else sMemI[n0 - HIDDEN] = s0;
        if (n1 < HIDDEN) sTau[n1] = s1; else sMemI[n1 - HIDDEN] = s1;
      }
    }
    __syncthreads();
    // --- In-place float4 blend of the delay line + next-step staging.
    // Physical slot p holds logical j = (p-base-1)&127 after the virtual
    // shift; the old head slot is the zero pad (zeroed by its owner thread,
    // thread-private). j/128 computed as exact fp32 fract (all values k/128).
    // buf = (1-iw)*shifted + iw*msg,  iw = memi*(1-|tau-j/128|)^2.
    {
      int h  = tid >> 1;
      int p0 = (tid & 1) << 6;
      float tau = sTau[h], memi = sMemI[h], m = sMsg[h];
      float* row = sBuf + h * MAX_DELAY;
      int zp = base & (MAX_DELAY - 1);
      if (zp >= p0 && zp < p0 + 64) row[zp] = 0.0f;
      float u0 = (float)(p0 - base - 1) * (1.0f / (float)MAX_DELAY);
#pragma unroll
      for (int q = 0; q < 64; q += 4) {
        v4f old = *(v4f*)(row + p0 + q);
        v4f nw;
#pragma unroll
        for (int e = 0; e < 4; ++e) {
          float t  = u0 + (float)(q + e) * (1.0f / (float)MAX_DELAY);
          float u  = t - floorf(t);                 // = j/128, exact
          float d  = 1.0f - fabsf(tau - u);
          float iw = memi * d * d;
          nw[e] = old[e] + iw * (m - old[e]);
        }
        *(v4f*)(row + p0 + q) = nw;
      }
      // Stage the next step's activation vector: new head = slot (base+1)&127
      // of the freshly blended row (this thread owns it), plus x row / zeros.
      int zn = (base + 1) & (MAX_DELAY - 1);
      if (zn >= p0 && zn < p0 + 64) a16[INPUT + h] = (_Float16)row[zn];
      if (tid < INPUT) a16[tid] = xnext ? (_Float16)xnext[tid] : (_Float16)0.0f;
    }
    __syncthreads();
  };

  // ---------------- Encoder: masked steps are exact no-ops -> run len steps.
  for (int t = 0; t < len; ++t) {
    const float* xn = (t + 1 < len) ? &x[((size_t)b * TLEN + t + 1) * INPUT]
                                    : nullptr;   // zeros = decoder's x0
    step(xn, false, 0);
    ++base;
  }

  // ---------------- Decoder: 64 steps, emit msg @ Wo + bo.
  for (int s = 0; s < OUT_LEN; ++s) {
    step(nullptr, true, s);
    ++base;
  }
}

extern "C" void kernel_launch(void* const* d_in, const int* in_sizes, int n_in,
                              void* d_out, int out_size, void* d_ws, size_t ws_size,
                              hipStream_t stream) {
  (void)in_sizes; (void)n_in; (void)out_size; (void)d_ws; (void)ws_size;
  const float* x       = (const float*)d_in[0];
  const int*   lengths = (const int*)d_in[1];
  // d_in[2] = out_lengths scalar (compile-time OUT_LEN = 64)
  const float* Wm = (const float*)d_in[3];
  const float* bm = (const float*)d_in[4];
  const float* W1 = (const float*)d_in[5];
  const float* b1 = (const float*)d_in[6];
  const float* W2 = (const float*)d_in[7];
  const float* b2 = (const float*)d_in[8];
  const float* Wo = (const float*)d_in[9];
  const float* bo = (const float*)d_in[10];
  float* out = (float*)d_out;

  // ~66 KB dynamic LDS per block (gfx1250 WGP has 320 KB).
  (void)hipFuncSetAttribute(reinterpret_cast<const void*>(delay_rnn_kernel),
                            hipFuncAttributeMaxDynamicSharedMemorySize, SMEM_BYTES);
  delay_rnn_kernel<<<dim3(BATCH), dim3(NTHREADS), SMEM_BYTES, stream>>>(
      x, lengths, Wm, bm, W1, b1, W2, b2, Wo, bo, out);
}
